// SGRUCell_2757369004590
// MI455X (gfx1250) — matrix operand order repaired and checked by reading, not compiled, not run on hardware
//
#include <hip/hip_runtime.h>
#include <hip/hip_bf16.h>
#include <stdint.h>

// ---------------------------------------------------------------------------
// SGRU scan for MI455X (gfx1250, wave32, WMMA).
// B=128, T=256, U=1024, NC=5000, TD=128.
//
// Design: fold the base GEMMs (K=128 _d and K=128 _s projections) into the
// recurrent GEMM K dimension -> per phase: (128 x 1280) @ (1280 x N) with
// N = 2048 (zr) / 1024 (hh) / 1024 (o). Weights pre-transposed+bf16 (10.5 MB,
// L2-resident). f32 accumulate, f32 state, bf16 WMMA 16x16x32.
// ---------------------------------------------------------------------------

#define B_   128
#define T_   256
#define U_   1024
#define K_   1280   // 1024 (h) + 128 (_d) + 128 (_s)
#define N4U  4096

typedef __attribute__((ext_vector_type(16))) __bf16 v16bf;
typedef __attribute__((ext_vector_type(8)))  float  v8f;

union FragU {
    uint4  q[2];
    v16bf  v;
};

__device__ __forceinline__ unsigned short f2bf(float f) {
    unsigned int u = __float_as_uint(f);
    unsigned int r = u + 0x7FFFu + ((u >> 16) & 1u);   // round-to-nearest-even
    return (unsigned short)(r >> 16);
}

__device__ __forceinline__ float sigmoidf_(float x) {
    return 1.0f / (1.0f + __expf(-x));
}

// ---------------------------------------------------------------------------
// zero a float buffer
// ---------------------------------------------------------------------------
__global__ void zero_f32(float* __restrict__ p, int n) {
    int i = blockIdx.x * blockDim.x + threadIdx.x;
    if (i < n) p[i] = 0.0f;
}

// ---------------------------------------------------------------------------
// Wt[n][k] (bf16, n-major, k contiguous) = concat([recurrent_kernel,
// kernel_d, kernel_s]) transposed.  blockIdx.x = k (0..1279)
// ---------------------------------------------------------------------------
__global__ __launch_bounds__(256) void prep_wt(
    const float* __restrict__ rk,   // (1024, 4096)
    const float* __restrict__ kd,   // (128, 4096)
    const float* __restrict__ ks,   // (128, 4096)
    unsigned short* __restrict__ Wt)
{
    int k = blockIdx.x;
    const float* src;
    if      (k < 1024) src = rk + (size_t)k * N4U;
    else if (k < 1152) src = kd + (size_t)(k - 1024) * N4U;
    else               src = ks + (size_t)(k - 1152) * N4U;
    for (int n = threadIdx.x; n < N4U; n += 256)
        Wt[(size_t)n * K_ + k] = f2bf(src[n]);
}

// ---------------------------------------------------------------------------
// Per-batch prefix-sum of x[:,:,0:2], then _d/_s tanh projections (bf16),
// stored t-major: dbuf[(t*B + b)*128 + j].  blockIdx.x = b, threadIdx.x = j.
// ---------------------------------------------------------------------------
__global__ __launch_bounds__(128) void prep_ds(
    const float* __restrict__ x,    // (B, T, 5)
    const float* __restrict__ Wd,   // (2, 128)
    const float* __restrict__ bd,   // (128)
    const float* __restrict__ Ws,   // (3, 128)
    const float* __restrict__ bs,   // (128)
    unsigned short* __restrict__ dbuf,
    unsigned short* __restrict__ sbuf)
{
    const int b = blockIdx.x;
    const int j = threadIdx.x;
    const float wd0 = Wd[j], wd1 = Wd[128 + j];
    const float ws0 = Ws[j], ws1 = Ws[128 + j], ws2 = Ws[256 + j];
    const float bdj = bd[j], bsj = bs[j];
    float d0 = 0.0f, d1 = 0.0f;
    for (int t = 0; t < T_; ++t) {
        const float* xp = x + ((size_t)b * T_ + t) * 5;
        d0 += xp[0];
        d1 += xp[1];
        float vd = tanhf(d0 * wd0 + d1 * wd1 + bdj);
        float vs = tanhf(xp[2] * ws0 + xp[3] * ws1 + xp[4] * ws2 + bsj);
        size_t idx = ((size_t)t * B_ + b) * 128 + j;
        dbuf[idx] = f2bf(vd);
        sbuf[idx] = f2bf(vs);
    }
}

// ---------------------------------------------------------------------------
// One GEMM phase of one timestep.
//   PHASE 1: A=[h,_d,_s]       -> zrpre[:, 0:2048]       (raw pre-activation)
//   PHASE 2: A=[r*h,_d,_s]     -> hhpre[:, 0:1024]       (raw pre-activation)
//   PHASE 3: A=[h_new,_d,_s]   -> out[b,t,:] = tanh(.+cb); writes h_new->hout
// Block = 256 threads (8 waves). Block tile = 16 (M) x 128 (N).
// gridDim = (N/128, 8).
// ---------------------------------------------------------------------------
template <int PHASE>
__global__ __launch_bounds__(256) void sgru_phase(
    const float* __restrict__ hin,
    float* __restrict__ hout,
    const unsigned short* __restrict__ Wt,      // (4096, 1280) bf16
    const unsigned short* __restrict__ dbuf,
    const unsigned short* __restrict__ sbuf,
    float* __restrict__ zrpre,                  // (128, 2048)
    float* __restrict__ hhpre,                  // (128, 1024)
    const int* __restrict__ ch,                 // (B, T)
    const float* __restrict__ kc,               // (5000, 4096)
    const float* __restrict__ bias,             // (4096)
    float* __restrict__ out,                    // (B, T, 1024)
    int t)
{
    __shared__ __align__(16) unsigned short As[16 * K_];   // 40 KB

    const int tid   = threadIdx.x;
    const int m0    = blockIdx.y * 16;
    const int nbase = (PHASE == 1) ? 0 : ((PHASE == 2) ? 2048 : 3072);
    const int n0    = nbase + blockIdx.x * 128;

    // ---- prologue: build bf16 A tile (gate elementwise math fused here) ----
    for (int idx = tid; idx < 16 * K_; idx += 256) {
        const int m = idx / K_;
        const int k = idx - m * K_;
        const int b = m0 + m;
        if (k >= 1024) {
            unsigned short raw = (k < 1152)
                ? dbuf[((size_t)t * B_ + b) * 128 + (k - 1024)]
                : sbuf[((size_t)t * B_ + b) * 128 + (k - 1152)];
            As[idx] = raw;
            continue;
        }
        float v;
        if (PHASE == 1) {
            v = hin[(size_t)b * U_ + k];
        } else if (PHASE == 2) {
            const size_t crow = (size_t)ch[b * T_ + t] * N4U;
            float cb = kc[crow + 1024 + k] + bias[1024 + k];
            float r  = sigmoidf_(zrpre[(size_t)b * 2048 + 1024 + k] + cb);
            v = r * hin[(size_t)b * U_ + k];
        } else {
            const size_t crow = (size_t)ch[b * T_ + t] * N4U;
            float cz  = kc[crow + k]        + bias[k];
            float chh = kc[crow + 2048 + k] + bias[2048 + k];
            float z   = sigmoidf_(zrpre[(size_t)b * 2048 + k] + cz);
            float hh  = tanhf(hhpre[(size_t)b * U_ + k] + chh);
            float h   = hin[(size_t)b * U_ + k];
            v = z * h + (1.0f - z) * hh;
            if (blockIdx.x == 0) hout[(size_t)b * U_ + k] = v;  // unique writer
        }
        As[idx] = f2bf(v);
    }
    __syncthreads();

    // ---- main K loop: 40 x v_wmma_f32_16x16x32_bf16 per wave --------------
    const int lane = tid & 31;
    const int wave = tid >> 5;
    const int rc   = lane & 15;   // A row / B,C column within tile
    const int g    = lane >> 4;   // half-wave selector
    const int ncol = n0 + wave * 16 + rc;           // global output column
    const unsigned short* wrow = Wt + (size_t)ncol * K_;

    v8f acc = {0.f, 0.f, 0.f, 0.f, 0.f, 0.f, 0.f, 0.f};

    for (int k0 = 0; k0 < K_; k0 += 32) {
        FragU a, bm;
        // A fragment: lane holds row rc, K = {k0+g*8..+7} U {k0+16+g*8..+7}
        a.q[0] = *reinterpret_cast<const uint4*>(&As[rc * K_ + k0 + g * 8]);
        a.q[1] = *reinterpret_cast<const uint4*>(&As[rc * K_ + k0 + 16 + g * 8]);
        // B fragment: lane holds column ncol, K = {k0+g*8..+7} U {k0+16+g*8..+7}
        bm.q[0] = *reinterpret_cast<const uint4*>(wrow + k0 + g * 8);
        bm.q[1] = *reinterpret_cast<const uint4*>(wrow + k0 + 16 + g * 8);
        __builtin_prefetch((const void*)(wrow + k0 + 256), 0, 1);
        acc = __builtin_amdgcn_wmma_f32_16x16x32_bf16(
            false, a.v, false, bm.v, (short)0, acc, false, false);
    }

    // ---- epilogue: C layout  vgpr i -> M = i + 8*g, N = lane&15 -----------
    for (int i = 0; i < 8; ++i) {
        const int b = m0 + i + g * 8;
        const float val = acc[i];
        if (PHASE == 1) {
            zrpre[(size_t)b * 2048 + ncol] = val;
        } else if (PHASE == 2) {
            hhpre[(size_t)b * U_ + (ncol - 2048)] = val;
        } else {
            float cb = kc[(size_t)ch[b * T_ + t] * N4U + ncol] + bias[ncol];
            out[((size_t)b * T_ + t) * U_ + (ncol - 3072)] = tanhf(val + cb);
        }
    }
}

// ---------------------------------------------------------------------------
// host side
// ---------------------------------------------------------------------------
extern "C" void kernel_launch(void* const* d_in, const int* in_sizes, int n_in,
                              void* d_out, int out_size, void* d_ws, size_t ws_size,
                              hipStream_t stream) {
    const float* x    = (const float*)d_in[0];
    const int*   ch   = (const int*)  d_in[1];
    const float* Wd   = (const float*)d_in[2];
    const float* bd   = (const float*)d_in[3];
    const float* Ws   = (const float*)d_in[4];
    const float* bs   = (const float*)d_in[5];
    const float* rk   = (const float*)d_in[6];
    const float* kc   = (const float*)d_in[7];
    const float* kd   = (const float*)d_in[8];
    const float* ks   = (const float*)d_in[9];
    const float* bias = (const float*)d_in[10];
    float* out = (float*)d_out;

    // workspace layout (all chunks 256B-aligned)
    uint8_t* w = (uint8_t*)d_ws;
    unsigned short* Wt   = (unsigned short*)w;               // 4096*1280*2 = 10,485,760
    unsigned short* dbuf = (unsigned short*)(w + 10485760);  // 256*128*128*2 =  8,388,608
    unsigned short* sbuf = (unsigned short*)(w + 18874368);  //                  8,388,608
    float* hA    = (float*)(w + 27262976);                   // 128*1024*4 =       524,288
    float* hB    = (float*)(w + 27787264);                   //                    524,288
    float* zrpre = (float*)(w + 28311552);                   // 128*2048*4 =     1,048,576
    float* hhpre = (float*)(w + 29360128);                   //                    524,288
    (void)ws_size; (void)in_sizes; (void)n_in; (void)out_size;

    // prep: zero h0, transpose/quantize weights, time-parallel _d/_s
    zero_f32<<<(B_ * U_ + 255) / 256, 256, 0, stream>>>(hA, B_ * U_);
    prep_wt<<<K_, 256, 0, stream>>>(rk, kd, ks, Wt);
    prep_ds<<<B_, 128, 0, stream>>>(x, Wd, bd, Ws, bs, dbuf, sbuf);

    float* hbufs[2] = {hA, hB};
    for (int t = 0; t < T_; ++t) {
        const float* hin = hbufs[t & 1];
        float*       hout = hbufs[(t + 1) & 1];
        sgru_phase<1><<<dim3(16, 8), 256, 0, stream>>>(
            hin, hout, Wt, dbuf, sbuf, zrpre, hhpre, ch, kc, bias, out, t);
        sgru_phase<2><<<dim3(8, 8), 256, 0, stream>>>(
            hin, hout, Wt, dbuf, sbuf, zrpre, hhpre, ch, kc, bias, out, t);
        sgru_phase<3><<<dim3(8, 8), 256, 0, stream>>>(
            hin, hout, Wt, dbuf, sbuf, zrpre, hhpre, ch, kc, bias, out, t);
    }
}
